// LSTMnet_46626164965898
// MI455X (gfx1250) — matrix operand order repaired and checked by reading, not compiled, not run on hardware
//
#include <hip/hip_runtime.h>

// ---------------------------------------------------------------------------
// Problem sizes (compile-time constants from the reference)
// ---------------------------------------------------------------------------
#define BATCH 64
#define SEQ   512
#define EMBD  512
#define HID   512
#define GATES 2048   // 4*HID, PyTorch gate order i,f,g,o
#define RDIM  100

typedef __attribute__((ext_vector_type(16))) __bf16 bf16x16;
typedef __attribute__((ext_vector_type(8)))  float  floatx8;

union Frag {            // one WMMA 16-bf16 operand fragment (8 VGPRs)
    bf16x16 v;
    uint4   u[2];
};
union Frag8 {           // 8 bf16 = 16 bytes (one b128)
    __bf16 e[8];
    uint4  u;
};

__device__ __forceinline__ floatx8 vzero8() {
    floatx8 z;
#pragma unroll
    for (int i = 0; i < 8; ++i) z[i] = 0.0f;
    return z;
}

__device__ __forceinline__ float sigf(float v) {
    return 1.0f / (1.0f + __expf(-v));
}

__device__ __forceinline__ floatx8 wmma_bf16(const Frag &a, const Frag &b, floatx8 c) {
    // D = A(16x32 bf16) * B(32x16 bf16) + C(16x16 f32)
    return __builtin_amdgcn_wmma_f32_16x16x32_bf16(
        false, a.v, false, b.v, (short)0, c, false, false);
}

// ---------------------------------------------------------------------------
// Kernel 0: fp32 -> bf16 weight conversion (W_ih and W_hh, both [2048,512]
// row-major, K-contiguous => already in WMMA-B-friendly layout).
// ---------------------------------------------------------------------------
__global__ __launch_bounds__(256) void cvt_weights_kernel(
    const float *__restrict__ wih, const float *__restrict__ whh,
    __bf16 *__restrict__ out_ih, __bf16 *__restrict__ out_hh) {
    size_t stride = (size_t)gridDim.x * blockDim.x;
    for (size_t i = (size_t)blockIdx.x * blockDim.x + threadIdx.x;
         i < (size_t)GATES * EMBD; i += stride) {
        out_ih[i] = (__bf16)wih[i];
        out_hh[i] = (__bf16)whh[i];
    }
}

// ---------------------------------------------------------------------------
// Kernel 1: xg = emb[x] @ W_ih^T + (b_ih + b_hh), bf16, stored TILE-MAJOR:
//   xg[((t*4 + bq)*128 + ntile)*256 + lane*8 + r]
// so each 16x16 C tile stays in its native WMMA lane layout and the
// recurrent kernel re-loads it with one b128 per lane per tile.
// GEMM M=T*B=32768 (gathered rows), N=2048, K=512; one block per M-tile;
// 16 waves x 8 N-tiles. B fragments double-buffered across k-steps with
// sched_barrier fences so loads stay in flight during the WMMA groups.
// ---------------------------------------------------------------------------
#define AP 520   // LDS pitch (bf16) for the 16x512 gathered A panel

__global__ __launch_bounds__(512) void xg_gemm_kernel(
    const int *__restrict__ x, const float *__restrict__ emb,
    const __bf16 *__restrict__ Wih, const float *__restrict__ b_ih,
    const float *__restrict__ b_hh, __bf16 *__restrict__ xg) {
    __shared__ alignas(16) __bf16 A_lds[16 * AP];

    const int tid  = threadIdx.x;
    const int wave = tid >> 5;        // 0..15, owns n-tiles [8w, 8w+8)
    const int lane = tid & 31;
    const int lm   = lane & 15;
    const int hi   = lane >> 4;

    const int m0 = blockIdx.x * 16;   // m = t*64 + b
    const int t  = m0 >> 6;
    const int b0 = m0 & 63;
    const int bq = b0 >> 4;

    // Gather 16 embedding rows, convert to bf16 into LDS.
    for (int i = tid; i < 16 * EMBD; i += 512) {
        const int row = i >> 9;
        const int k   = i & (EMBD - 1);
        const int idx = x[(b0 + row) * SEQ + t];
        A_lds[row * AP + k] = (__bf16)emb[(size_t)idx * EMBD + k];
    }
    __syncthreads();

    floatx8 acc[8];
#pragma unroll
    for (int nt = 0; nt < 8; ++nt) acc[nt] = vzero8();

    const __bf16 *wrow = Wih + (size_t)(wave * 128 + lm) * EMBD;

    auto loadB = [&](Frag *b, int kk) {
        const int kloc = kk * 32 + 16 * hi;
#pragma unroll
        for (int i = 0; i < 8; ++i) {
            const __bf16 *bp = wrow + (size_t)i * 16 * EMBD + kloc;
            b[i].u[0] = *(const uint4 *)bp;
            b[i].u[1] = *(const uint4 *)(bp + 8);
        }
    };
    auto loadA = [&](Frag &a, int kk) {
        const __bf16 *ap = &A_lds[lm * AP + kk * 32 + 16 * hi];
        a.u[0] = *(const uint4 *)ap;          // K = kk*32 + 16*hi .. +7
        a.u[1] = *(const uint4 *)(ap + 8);    // K = kk*32 + 16*hi + 8 .. +15
    };

    Frag b0f[8], b1f[8], a0, a1;
    loadB(b0f, 0);
    for (int kk = 0; kk < EMBD / 32; kk += 2) {
        loadA(a0, kk);
        loadA(a1, kk + 1);
        loadB(b1f, kk + 1);
        __builtin_amdgcn_sched_barrier(0);
#pragma unroll
        for (int i = 0; i < 8; ++i) acc[i] = wmma_bf16(a0, b0f[i], acc[i]);
        if (kk + 2 < EMBD / 32) loadB(b0f, kk + 2);
        __builtin_amdgcn_sched_barrier(0);
#pragma unroll
        for (int i = 0; i < 8; ++i) acc[i] = wmma_bf16(a1, b1f[i], acc[i]);
    }

#pragma unroll
    for (int nt = 0; nt < 8; ++nt) {
        const int ntg  = wave * 8 + nt;          // global n-tile 0..127
        const int n    = ntg * 16 + lm;
        const float bv = b_ih[n] + b_hh[n];
        Frag8 o;
#pragma unroll
        for (int r = 0; r < 8; ++r) o.e[r] = (__bf16)(acc[nt][r] + bv);
        *(uint4 *)(xg + ((((size_t)t * 4 + bq) * 128 + ntg) << 8) + lane * 8) = o.u;
    }
}

// ---------------------------------------------------------------------------
// Kernel 2: the sequential LSTM recurrence. 4 blocks x 16 batch rows each
// (batch rows recur independently -> NO inter-workgroup sync). 32 waves per
// block (8/SIMD); wave w owns hidden units j in [16w, 16w+16) = gate columns
// {j, 512+j, 1024+j, 1536+j} (one WMMA N-tile per gate) -> c/h update is
// register math in the WMMA C layout; c persists in one v8f per lane for all
// 512 steps. h round-trips via a padded LDS panel; W_hh streams from the
// 192MB L2 every step. The k-loop is software-pipelined: B (and A) fragments
// for k+1 are issued before a sched_barrier fence, WMMAs for k run after it,
// so per-wave loads overlap WMMA execution instead of s_wait_loadcnt 0x0.
// The xg C-init b128s are issued before the k-loop and converted only at the
// gate stage, hiding their latency entirely.
// USE_XG=1: input contribution precomputed (K=512/step).
// USE_XG=0: fused fallback, stage embeddings per step (K=1024/step).
// ---------------------------------------------------------------------------
#define KP 1032  // LDS panel pitch (bf16): 16 rows x (e[0:512] | h[512:1024])

template <int USE_XG>
__global__ __launch_bounds__(1024) void lstm_rec_kernel(
    const int *__restrict__ x, const float *__restrict__ emb,
    const __bf16 *__restrict__ Wih, const __bf16 *__restrict__ Whh,
    const float *__restrict__ b_ih, const float *__restrict__ b_hh,
    const __bf16 *__restrict__ xg, float *__restrict__ a_out) {
    __shared__ alignas(16) __bf16 panel[16 * KP];

    const int tid  = threadIdx.x;
    const int wave = tid >> 5;                 // 0..31
    const int lane = tid & 31;
    const int lm   = lane & 15;
    const int hi   = lane >> 4;
    const int bs   = blockIdx.x * 16;          // batch slice [bs, bs+16)
    const int bq   = blockIdx.x;
    const int hoff = USE_XG ? 0 : 512;         // h columns in the panel
    constexpr int NKK = USE_XG ? (HID / 32) : ((EMBD + HID) / 32);
    const int j    = wave * 16 + lm;           // this lane's hidden column

    // h0 = 0 everywhere; stage e_0 if fused.
    for (int i = tid; i < 16 * KP; i += 1024) panel[i] = (__bf16)0.0f;
    if (!USE_XG) {
        for (int i = tid; i < 16 * EMBD; i += 1024) {
            const int row = i >> 9, k = i & (EMBD - 1);
            const int idx = x[(bs + row) * SEQ + 0];
            panel[row * KP + k] = (__bf16)emb[(size_t)idx * EMBD + k];
        }
    }
    __syncthreads();

    // Loop-invariant per-gate state: column, weight-row base, bias.
    const __bf16 *wb_hh[4];
    const __bf16 *wb_ih[4];
    float biasv[4];
#pragma unroll
    for (int tau = 0; tau < 4; ++tau) {
        const int n = tau * HID + j;
        wb_hh[tau]  = Whh + (size_t)n * HID;
        wb_ih[tau]  = Wih + (size_t)n * EMBD;
        biasv[tau]  = USE_XG ? 0.0f : (b_ih[n] + b_hh[n]);
    }

    auto loadB = [&](Frag *b, int kk) {
        const int  kbase  = kk * 32 + 16 * hi;
        const bool use_hh = USE_XG || (kbase >= EMBD);
        const int  kloc   = (USE_XG || !use_hh) ? kbase : (kbase - EMBD);
#pragma unroll
        for (int tau = 0; tau < 4; ++tau) {
            const __bf16 *bp = (use_hh ? wb_hh[tau] : wb_ih[tau]) + kloc;
            b[tau].u[0] = *(const uint4 *)bp;
            b[tau].u[1] = *(const uint4 *)(bp + 8);
        }
    };
    auto loadA = [&](Frag &a, int kk) {
        const __bf16 *ap = &panel[lm * KP + kk * 32 + 16 * hi];
        a.u[0] = *(const uint4 *)ap;
        a.u[1] = *(const uint4 *)(ap + 8);
    };

    floatx8 c = vzero8();

    for (int t = 0; t < SEQ; ++t) {
        // Issue the xg C-init loads up front; convert only at the gate stage.
        uint4 xraw[4];
        if (USE_XG) {
            const size_t tb = ((size_t)t * 4 + bq) * 128;
#pragma unroll
            for (int tau = 0; tau < 4; ++tau)
                xraw[tau] = *(const uint4 *)(
                    xg + ((tb + (size_t)tau * 32 + wave) << 8) + lane * 8);
        }

        floatx8 acc[4];
#pragma unroll
        for (int tau = 0; tau < 4; ++tau) acc[tau] = vzero8();

        Frag b0f[4], b1f[4], a0, a1;
        loadB(b0f, 0);
        for (int kk = 0; kk < NKK; kk += 2) {
            loadA(a0, kk);
            loadA(a1, kk + 1);
            loadB(b1f, kk + 1);
            __builtin_amdgcn_sched_barrier(0);
#pragma unroll
            for (int tau = 0; tau < 4; ++tau)
                acc[tau] = wmma_bf16(a0, b0f[tau], acc[tau]);
            if (kk + 2 < NKK) loadB(b0f, kk + 2);
            __builtin_amdgcn_sched_barrier(0);
#pragma unroll
            for (int tau = 0; tau < 4; ++tau)
                acc[tau] = wmma_bf16(a1, b1f[tau], acc[tau]);
        }

        // Gate fusion in the C layout: lane holds rows m = r+8*hi of col j.
        floatx8 hv;
        {
            Frag8 xv[4];
#pragma unroll
            for (int tau = 0; tau < 4; ++tau) xv[tau].u = xraw[tau];
#pragma unroll
            for (int r = 0; r < 8; ++r) {
                const float a0v = USE_XG ? (float)xv[0].e[r] : biasv[0];
                const float a1v = USE_XG ? (float)xv[1].e[r] : biasv[1];
                const float a2v = USE_XG ? (float)xv[2].e[r] : biasv[2];
                const float a3v = USE_XG ? (float)xv[3].e[r] : biasv[3];
                const float gi = sigf(acc[0][r] + a0v);
                const float gf = sigf(acc[1][r] + a1v);
                const float gg = tanhf(acc[2][r] + a2v);
                const float go = sigf(acc[3][r] + a3v);
                const float cn = gf * c[r] + gi * gg;
                c[r]  = cn;
                hv[r] = go * tanhf(cn);
            }
        }

        __syncthreads();   // all waves finished reading the panel

#pragma unroll
        for (int r = 0; r < 8; ++r)
            panel[(r + 8 * hi) * KP + hoff + j] = (__bf16)hv[r];

        if (!USE_XG && t + 1 < SEQ) {   // stage e_{t+1}
            for (int i = tid; i < 16 * EMBD; i += 1024) {
                const int row = i >> 9, k = i & (EMBD - 1);
                const int idx = x[(bs + row) * SEQ + (t + 1)];
                panel[row * KP + k] = (__bf16)emb[(size_t)idx * EMBD + k];
            }
        }
        if (t == SEQ - 1) {             // a = relu(h_T), fp32
#pragma unroll
            for (int r = 0; r < 8; ++r) {
                const float hvv = hv[r];
                a_out[(size_t)(bs + r + 8 * hi) * HID + j] = hvv > 0.0f ? hvv : 0.0f;
            }
        }
        __syncthreads();   // panel updated for step t+1
    }
}

// ---------------------------------------------------------------------------
// Kernel 3: head.  doc = a @ W_doc^T + b_doc  (64x100, K=512);
//                  logits[b] = doc[b] . W_out + b_out.  ~7 MFLOP -> plain f32.
// ---------------------------------------------------------------------------
__global__ __launch_bounds__(256) void head_kernel(
    const float *__restrict__ a, const float *__restrict__ Wdoc,
    const float *__restrict__ bdoc, const float *__restrict__ Wout,
    const float *__restrict__ bout, float *__restrict__ out) {
    __shared__ float doc[BATCH * RDIM];
    const int tid = threadIdx.x;
    for (int i = tid; i < BATCH * RDIM; i += 256) {
        const int b = i / RDIM, r = i % RDIM;
        const float *ap = a + (size_t)b * HID;
        const float *wp = Wdoc + (size_t)r * HID;
        float s = bdoc[r];
        for (int k = 0; k < HID; ++k) s += ap[k] * wp[k];
        doc[i] = s;
    }
    __syncthreads();
    if (tid < BATCH) {
        float s = bout[0];
        for (int r = 0; r < RDIM; ++r) s += doc[tid * RDIM + r] * Wout[r];
        out[tid] = s;
    }
}

// ---------------------------------------------------------------------------
// Launch.  Workspace layout (bytes):
//   [0, 2M)        W_ih  bf16  [2048][512]
//   [2M, 4M)       W_hh  bf16  [2048][512]
//   [4M, 4M+128K)  a = relu(h_T)  f32 [64][512]
//   [4M+128K, ...) xg bf16, tile-major [512][4][128][256] (134.2 MB),
//                  only if ws fits (checked at runtime; else fused mode).
// ---------------------------------------------------------------------------
extern "C" void kernel_launch(void *const *d_in, const int *in_sizes, int n_in,
                              void *d_out, int out_size, void *d_ws, size_t ws_size,
                              hipStream_t stream) {
    const int   *x     = (const int *)d_in[0];
    const float *emb   = (const float *)d_in[1];
    const float *W_ih  = (const float *)d_in[2];
    const float *W_hh  = (const float *)d_in[3];
    const float *b_ih  = (const float *)d_in[4];
    const float *b_hh  = (const float *)d_in[5];
    const float *W_doc = (const float *)d_in[6];
    const float *b_doc = (const float *)d_in[7];
    const float *W_out = (const float *)d_in[8];
    const float *b_out = (const float *)d_in[9];
    float *out = (float *)d_out;

    char *ws = (char *)d_ws;
    const size_t OFF_WHH = (size_t)GATES * EMBD * 2;             // 2 MB
    const size_t OFF_A   = OFF_WHH * 2;                          // 4 MB
    const size_t OFF_XG  = OFF_A + (size_t)BATCH * HID * 4;      // +128 KB
    const size_t NEED_XG = OFF_XG + (size_t)SEQ * BATCH * GATES * 2;

    __bf16 *Wih_bf = (__bf16 *)(ws);
    __bf16 *Whh_bf = (__bf16 *)(ws + OFF_WHH);
    float  *a_buf  = (float *)(ws + OFF_A);
    __bf16 *xg     = (__bf16 *)(ws + OFF_XG);

    const bool use_xg = (ws_size >= NEED_XG);

    cvt_weights_kernel<<<1024, 256, 0, stream>>>(W_ih, W_hh, Wih_bf, Whh_bf);

    if (use_xg) {
        xg_gemm_kernel<<<(SEQ * BATCH) / 16, 512, 0, stream>>>(
            x, emb, Wih_bf, b_ih, b_hh, xg);
        lstm_rec_kernel<1><<<BATCH / 16, 1024, 0, stream>>>(
            x, emb, Wih_bf, Whh_bf, b_ih, b_hh, xg, a_buf);
    } else {
        lstm_rec_kernel<0><<<BATCH / 16, 1024, 0, stream>>>(
            x, emb, Wih_bf, Whh_bf, b_ih, b_hh, xg, a_buf);
    }

    head_kernel<<<1, 256, 0, stream>>>(a_buf, W_doc, b_doc, W_out, b_out, out);
}
